// CrossAttentionLayer_36378372997687
// MI455X (gfx1250) — matrix operand order repaired and checked by reading, not compiled, not run on hardware
//
#include <hip/hip_runtime.h>
#include <hip/hip_bf16.h>

typedef __attribute__((ext_vector_type(16))) _Float16 v16h;
typedef __attribute__((ext_vector_type(8)))  _Float16 v8h;
typedef __attribute__((ext_vector_type(4)))  _Float16 v4h;
typedef __attribute__((ext_vector_type(8)))  float    v8f;
typedef __attribute__((ext_vector_type(4)))  int      v4i;

#define Bb 4
#define Ss 2048
#define Hh 2048
#define NHh 16
#define HDd 128

#if __has_builtin(__builtin_amdgcn_global_load_async_to_lds_b128) && \
    __has_builtin(__builtin_amdgcn_s_wait_asynccnt)
#define HAVE_ASYNC_LDS 1
#endif

// ---------------------------------------------------------------------------
// WMMA helpers (CDNA5 16x16x32 f16 -> f32)
// ---------------------------------------------------------------------------
__device__ __forceinline__ v8f wmma16(v16h a, v16h b, v8f c) {
  return __builtin_amdgcn_wmma_f32_16x16x32_f16(
      /*neg_a=*/false, a, /*neg_b=*/false, b,
      /*c_mod=*/(short)0, c, /*reuse_a=*/false, /*reuse_b=*/false);
}

// A fragment (16x32 f16, MxK). lane m=l&15; chunks K[k0..k0+7], K[k0+16..k0+23]
// with k0 = (l>=16 ? 8 : 0). base = tile origin, stride = row stride in halves.
__device__ __forceinline__ v16h load_frag_a(const _Float16* base, int stride) {
  const int lane = threadIdx.x & 31;
  const int row  = lane & 15;
  const int k0   = (lane >> 4) * 8;
  const _Float16* p = base + row * stride + k0;
  v8h lo = *(const v8h*)(p);
  v8h hi = *(const v8h*)(p + 16);
  v16h r;
  for (int i = 0; i < 8; ++i) { r[i] = lo[i]; r[i + 8] = hi[i]; }
  return r;
}

// B fragment (32x16 f16, KxN) stored as [n][k]. lane n=l&15; chunks K[k0..k0+7], K[k0+16..k0+23]
// with k0 = (l>=16 ? 8 : 0).
__device__ __forceinline__ v16h load_frag_b(const _Float16* base, int stride) {
  const int lane = threadIdx.x & 31;
  const int col  = lane & 15;
  const int k0   = (lane >> 4) * 8;
  const _Float16* p = base + col * stride + k0;
  v8h lo = *(const v8h*)(p);
  v8h hi = *(const v8h*)(p + 16);
  v16h r;
  for (int i = 0; i < 8; ++i) { r[i] = lo[i]; r[i + 8] = hi[i]; }
  return r;
}

// 16-byte global -> LDS copy: async direct-to-LDS on CDNA5 (no VGPR round
// trip, tracked by ASYNCcnt), synchronous fallback otherwise.
__device__ __forceinline__ void cp16_g2l(const _Float16* g, _Float16* l) {
#if defined(HAVE_ASYNC_LDS)
  __builtin_amdgcn_global_load_async_to_lds_b128((v4i*)g, (v4i*)l, 0, 0);
#else
  *(v8h*)l = *(const v8h*)g;
#endif
}

__device__ __forceinline__ void wait_async_le4() {
#if defined(HAVE_ASYNC_LDS)
  __builtin_amdgcn_s_wait_asynccnt(4);
#endif
}

// ---------------------------------------------------------------------------
// LayerNorm (fp32 in, f16 out), one block per row of 2048
// ---------------------------------------------------------------------------
__global__ __launch_bounds__(256) void ln_f16_kernel(
    const float* __restrict__ x, const float* __restrict__ gamma,
    const float* __restrict__ beta, _Float16* __restrict__ out) {
  __shared__ float s1[8], s2[8];
  const int row = blockIdx.x, tid = threadIdx.x;
  const int wv = tid >> 5, lane = tid & 31;
  const float4* xr = (const float4*)(x + (size_t)row * Hh);
  float4 v0 = xr[tid], v1 = xr[tid + 256];
  float sum = v0.x + v0.y + v0.z + v0.w + v1.x + v1.y + v1.z + v1.w;
  float sq  = v0.x*v0.x + v0.y*v0.y + v0.z*v0.z + v0.w*v0.w
            + v1.x*v1.x + v1.y*v1.y + v1.z*v1.z + v1.w*v1.w;
  for (int off = 16; off >= 1; off >>= 1) {
    sum += __shfl_xor(sum, off, 32);
    sq  += __shfl_xor(sq, off, 32);
  }
  if (lane == 0) { s1[wv] = sum; s2[wv] = sq; }
  __syncthreads();
  float S = 0.f, Q = 0.f;
  for (int i = 0; i < 8; ++i) { S += s1[i]; Q += s2[i]; }
  const float mu  = S * (1.0f / Hh);
  const float var = Q * (1.0f / Hh) - mu * mu;
  const float inv = rsqrtf(var + 1e-5f);
  const float4* gr = (const float4*)gamma;
  const float4* br = (const float4*)beta;
  v4h* o4 = (v4h*)(out + (size_t)row * Hh);
  for (int i = 0; i < 2; ++i) {
    const int idx = tid + 256 * i;
    float4 g = gr[idx], bt = br[idx];
    float4 xv = i ? v1 : v0;
    v4h o;
    o[0] = (_Float16)((xv.x - mu) * inv * g.x + bt.x);
    o[1] = (_Float16)((xv.y - mu) * inv * g.y + bt.y);
    o[2] = (_Float16)((xv.z - mu) * inv * g.z + bt.z);
    o[3] = (_Float16)((xv.w - mu) * inv * g.w + bt.w);
    o4[idx] = o;
  }
}

// ---------------------------------------------------------------------------
// fp32 -> f16 elementwise (n4 = elements/4)
// ---------------------------------------------------------------------------
__global__ __launch_bounds__(256) void cvt_f16_kernel(
    const float* __restrict__ in, _Float16* __restrict__ out, int n4) {
  int i = blockIdx.x * blockDim.x + threadIdx.x;
  if (i < n4) {
    float4 v = ((const float4*)in)[i];
    v4h o;
    o[0] = (_Float16)v.x; o[1] = (_Float16)v.y;
    o[2] = (_Float16)v.z; o[3] = (_Float16)v.w;
    ((v4h*)out)[i] = o;
  }
}

// ---------------------------------------------------------------------------
// Tiled WMMA GEMM: C[M,N] = A[M,K] @ W[N,K]^T + bias
// Block tile 128x128, K-step 32, 8 waves each computing 32x64.
// Double-buffered LDS, async global->LDS prefetch of the next K-tile
// overlapped with the WMMAs of the current tile.
// mode 0: f16 out, Q/K head layout [b,h,s,d]
// mode 1: f16 out, V transposed   [b,h,d,s]
// mode 3: f32 out, d_out[row*N+col] = (acc+bias)*gate
// ---------------------------------------------------------------------------
__global__ __launch_bounds__(256) void gemm_wmma_kernel(
    const _Float16* __restrict__ A, const _Float16* __restrict__ W,
    const float* __restrict__ bias, void* __restrict__ out,
    const float* __restrict__ gate, int mode) {
  __shared__ _Float16 lds_a[2][128][40];
  __shared__ _Float16 lds_b[2][128][40];
  const int tid = threadIdx.x;
  const int wv = tid >> 5, lane = tid & 31;
  const int m0 = blockIdx.y * 128, n0 = blockIdx.x * 128;
  const int mw = (wv & 3) * 32, nw = (wv >> 2) * 64;

  v8f acc[2][4];
  for (int i = 0; i < 2; ++i)
    for (int j = 0; j < 4; ++j)
      for (int e = 0; e < 8; ++e) acc[i][j][e] = 0.f;

  const int lrow = tid >> 1;        // 0..127
  const int lk   = (tid & 1) * 16;  // 0 or 16

  // issue the 4 16-byte transfers staging K-tile kk into LDS buffer bufi
  auto stage = [&](int bufi, int kk) {
    const _Float16* ga = A + (size_t)(m0 + lrow) * Hh + kk + lk;
    cp16_g2l(ga,     &lds_a[bufi][lrow][lk]);
    cp16_g2l(ga + 8, &lds_a[bufi][lrow][lk + 8]);
    const _Float16* gb = W + (size_t)(n0 + lrow) * Hh + kk + lk;
    cp16_g2l(gb,     &lds_b[bufi][lrow][lk]);
    cp16_g2l(gb + 8, &lds_b[bufi][lrow][lk + 8]);
  };

  stage(0, 0);
  int buf = 0;
  for (int k = 0; k < Hh; k += 32) {
    // prefetch next tile (wraps on the last iteration; stays in-bounds and
    // keeps the asynccnt wait value a compile-time immediate)
    stage(buf ^ 1, (k + 32) & (Hh - 1));
    wait_async_le4();   // the 4 oldest transfers (current buffer) are done
    __syncthreads();
    v16h af[2];
    for (int i = 0; i < 2; ++i)
      af[i] = load_frag_a(&lds_a[buf][mw + 16 * i][0], 40);
    for (int j = 0; j < 4; ++j) {
      v16h bf = load_frag_b(&lds_b[buf][nw + 16 * j][0], 40);
      for (int i = 0; i < 2; ++i) acc[i][j] = wmma16(af[i], bf, acc[i][j]);
    }
    __syncthreads();
    buf ^= 1;
  }

  const int hi = lane >> 4, ln = lane & 15;
  const float g = (mode == 3) ? gate[0] : 0.f;
  for (int i = 0; i < 2; ++i)
    for (int j = 0; j < 4; ++j) {
      const int gmb = m0 + mw + 16 * i;
      const int col = n0 + nw + 16 * j + ln;
      const float bv = bias[col];
      const int h = col >> 7, d = col & (HDd - 1);
      for (int r = 0; r < 8; ++r) {
        const int row = gmb + r + hi * 8;
        const int b = row >> 11, s = row & (Ss - 1);
        const float val = acc[i][j][r] + bv;
        if (mode == 0) {
          ((_Float16*)out)[((size_t)(b * NHh + h) * Ss + s) * HDd + d] =
              (_Float16)val;
        } else if (mode == 1) {
          ((_Float16*)out)[((size_t)(b * NHh + h) * HDd + d) * Ss + s] =
              (_Float16)val;
        } else {
          ((float*)out)[(size_t)row * Hh + col] = val * g;
        }
      }
    }
}

// ---------------------------------------------------------------------------
// Flash attention: grid (S/128, NH, B), 8 waves, each wave = 16 query rows.
// scores = (Q @ K^T) / (SCALE * dyn); online softmax fp32; ctx = P @ V.
// Gate scale (gate*base+gate_bias) folded into epilogue. ctx: [b,s,h*HD+d] f16
// ---------------------------------------------------------------------------
__global__ __launch_bounds__(256) void attn_kernel(
    const _Float16* __restrict__ Q, const _Float16* __restrict__ K,
    const _Float16* __restrict__ Vt, _Float16* __restrict__ ctx,
    const float* __restrict__ gate, const float* __restrict__ gate_bias,
    const float* __restrict__ base_w, const float* __restrict__ dyn) {
  __shared__ _Float16 Plds[8][16][72];
  const int tid = threadIdx.x, wv = tid >> 5, lane = tid & 31;
  const int hi = lane >> 4, ln = lane & 15;
  const int b = blockIdx.z, h = blockIdx.y;
  const size_t bh = (size_t)(b * NHh + h);
  const _Float16* Qb = Q + bh * Ss * HDd;
  const _Float16* Kb = K + bh * Ss * HDd;
  const _Float16* Vb = Vt + bh * HDd * Ss;
  const int q0 = blockIdx.x * 128 + wv * 16;
  const float fac = 1.0f / (0.08838834764831843f * dyn[0]);  // *sqrt(HD)/dyn

  v16h qf[4];
  for (int i = 0; i < 4; ++i)
    qf[i] = load_frag_a(Qb + (size_t)q0 * HDd + 32 * i, HDd);

  v8f o[8];
  for (int j = 0; j < 8; ++j)
    for (int e = 0; e < 8; ++e) o[j][e] = 0.f;
  float mrow[8], lsum[8];
  for (int r = 0; r < 8; ++r) { mrow[r] = -1e30f; lsum[r] = 0.f; }

  for (int kv = 0; kv < Ss; kv += 64) {
    v8f sc[4];
    for (int t = 0; t < 4; ++t)
      for (int e = 0; e < 8; ++e) sc[t][e] = 0.f;
    for (int t = 0; t < 4; ++t)
      for (int i = 0; i < 4; ++i) {
        v16h bf = load_frag_b(Kb + (size_t)(kv + 16 * t) * HDd + 32 * i, HDd);
        sc[t] = wmma16(qf[i], bf, sc[t]);
      }
    // online softmax per row (rows r + hi*8; 16 lanes of a half share a row)
    for (int r = 0; r < 8; ++r) {
      float sr[4];
      float mx = -1e30f;
      for (int t = 0; t < 4; ++t) { sr[t] = sc[t][r] * fac; mx = fmaxf(mx, sr[t]); }
      for (int off = 8; off >= 1; off >>= 1) mx = fmaxf(mx, __shfl_xor(mx, off, 32));
      const float mnew  = fmaxf(mrow[r], mx);
      const float alpha = __expf(mrow[r] - mnew);
      float rs = 0.f;
      for (int t = 0; t < 4; ++t) {
        const float p = __expf(sr[t] - mnew);
        rs += p;
        Plds[wv][r + hi * 8][t * 16 + ln] = (_Float16)p;
      }
      for (int off = 8; off >= 1; off >>= 1) rs += __shfl_xor(rs, off, 32);
      lsum[r] = lsum[r] * alpha + rs;
      mrow[r] = mnew;
      for (int j = 0; j < 8; ++j) o[j][r] *= alpha;
    }
    __syncthreads();
    for (int i2 = 0; i2 < 2; ++i2) {
      v16h pa = load_frag_a(&Plds[wv][0][32 * i2], 72);
      for (int j = 0; j < 8; ++j) {
        v16h bv = load_frag_b(Vb + (size_t)(16 * j) * Ss + kv + 32 * i2, Ss);
        o[j] = wmma16(pa, bv, o[j]);
      }
    }
    __syncthreads();
  }

  const float gs = gate[0] * base_w[0] + gate_bias[0];
  for (int j = 0; j < 8; ++j)
    for (int r = 0; r < 8; ++r) {
      const int srow = q0 + r + hi * 8;
      const float val = o[j][r] * (gs / lsum[r]);
      ctx[((size_t)b * Ss + srow) * Hh + h * HDd + 16 * j + ln] = (_Float16)val;
    }
}

// ---------------------------------------------------------------------------
// Launch
// ---------------------------------------------------------------------------
extern "C" void kernel_launch(void* const* d_in, const int* in_sizes, int n_in,
                              void* d_out, int out_size, void* d_ws,
                              size_t ws_size, hipStream_t stream) {
  const float* hs    = (const float*)d_in[0];
  const float* cs    = (const float*)d_in[1];
  const float* Wq    = (const float*)d_in[2];
  const float* bq    = (const float*)d_in[3];
  const float* Wk    = (const float*)d_in[4];
  const float* bk    = (const float*)d_in[5];
  const float* Wv    = (const float*)d_in[6];
  const float* bv    = (const float*)d_in[7];
  const float* Wo    = (const float*)d_in[8];
  const float* bo    = (const float*)d_in[9];
  const float* lng   = (const float*)d_in[10];
  const float* lnb   = (const float*)d_in[11];
  const float* gate  = (const float*)d_in[12];
  const float* gbias = (const float*)d_in[13];
  const float* basew = (const float*)d_in[14];
  const float* dyn   = (const float*)d_in[15];

  char* ws = (char*)d_ws;
  size_t off = 0;
  auto take = [&](size_t bytes) -> char* {
    char* p = ws + off;
    off += (bytes + 255) & ~(size_t)255;
    return p;
  };
  const size_t act_bytes = (size_t)Bb * Ss * Hh * sizeof(_Float16);  // 32 MB
  const size_t w_bytes   = (size_t)Hh * Hh * sizeof(_Float16);       // 8 MB
  _Float16* ln16 = (_Float16*)take(act_bytes);
  _Float16* cr16 = (_Float16*)take(act_bytes);
  _Float16* wq16 = (_Float16*)take(w_bytes);
  _Float16* wk16 = (_Float16*)take(w_bytes);
  _Float16* wv16 = (_Float16*)take(w_bytes);
  _Float16* wo16 = (_Float16*)take(w_bytes);
  _Float16* Qh   = (_Float16*)take(act_bytes);
  _Float16* Kh   = (_Float16*)take(act_bytes);
  _Float16* VtT  = (_Float16*)take(act_bytes);
  _Float16* ctx  = ln16;  // safe alias: ln16 dead after Q projection

  // 1. LayerNorm -> f16
  ln_f16_kernel<<<Bb * Ss, 256, 0, stream>>>(hs, lng, lnb, ln16);
  // 2. conversions
  cvt_f16_kernel<<<(Bb * Ss * Hh / 4 + 255) / 256, 256, 0, stream>>>(
      cs, cr16, Bb * Ss * Hh / 4);
  cvt_f16_kernel<<<(Hh * Hh / 4 + 255) / 256, 256, 0, stream>>>(Wq, wq16, Hh * Hh / 4);
  cvt_f16_kernel<<<(Hh * Hh / 4 + 255) / 256, 256, 0, stream>>>(Wk, wk16, Hh * Hh / 4);
  cvt_f16_kernel<<<(Hh * Hh / 4 + 255) / 256, 256, 0, stream>>>(Wv, wv16, Hh * Hh / 4);
  cvt_f16_kernel<<<(Hh * Hh / 4 + 255) / 256, 256, 0, stream>>>(Wo, wo16, Hh * Hh / 4);
  // 3. projections (M = B*S = 8192, N = K = 2048)
  dim3 gg(Hh / 128, Bb * Ss / 128);
  gemm_wmma_kernel<<<gg, 256, 0, stream>>>(ln16, wq16, bq, Qh, nullptr, 0);
  gemm_wmma_kernel<<<gg, 256, 0, stream>>>(cr16, wk16, bk, Kh, nullptr, 0);
  gemm_wmma_kernel<<<gg, 256, 0, stream>>>(cr16, wv16, bv, VtT, nullptr, 1);
  // 4. flash attention
  dim3 ga(Ss / 128, NHh, Bb);
  attn_kernel<<<ga, 256, 0, stream>>>(Qh, Kh, VtT, ctx, gate, gbias, basew, dyn);
  // 5. output projection -> fp32 d_out, * gate
  gemm_wmma_kernel<<<gg, 256, 0, stream>>>(ctx, wo16, bo, d_out, gate, 3);
}
